// _TXCEncoderVariantBase_81612968558795
// MI455X (gfx1250) — compile-verified, matrix operand order checked by reading, not run on hardware
//
#include <hip/hip_runtime.h>

// ---------------------------------------------------------------------------
// TopK Sparse Autoencoder forward for MI455X (gfx1250, wave32, WMMA + TDM).
//   B=4096, T=8, D_IN=512  -> KD = T*D_IN = 4096
//   D_SAE=16384, TOPK=32
// Outputs (concatenated): [loss(1)] [x_hat(B*KD)] [z(B*D_SAE)]
// ---------------------------------------------------------------------------

typedef _Float16     v16h  __attribute__((ext_vector_type(16)));
typedef float        v8f   __attribute__((ext_vector_type(8)));
typedef unsigned int u32x4 __attribute__((ext_vector_type(4)));
typedef int          i32x4 __attribute__((ext_vector_type(4)));
typedef int          i32x8 __attribute__((ext_vector_type(8)));

constexpr int BATCH = 4096;
constexpr int KD    = 4096;     // T * D_IN
constexpr int DSAE  = 16384;
constexpr int TOPK  = 32;

#if defined(__AMDGCN__) && __has_builtin(__builtin_amdgcn_tensor_load_to_lds)
#define USE_TDM 1
#else
#define USE_TDM 0
#endif

// ---------------------------------------------------------------------------
// fp32 -> f16 conversion, 4 elements per thread
// ---------------------------------------------------------------------------
__global__ __launch_bounds__(256) void cvt_f32_f16(const float* __restrict__ in,
                                                   _Float16* __restrict__ out,
                                                   int n4) {
  int i = blockIdx.x * 256 + threadIdx.x;
  if (i >= n4) return;
  float4 f = reinterpret_cast<const float4*>(in)[i];
  union { ushort4 u; _Float16 h[4]; } o;
  o.h[0] = (_Float16)f.x; o.h[1] = (_Float16)f.y;
  o.h[2] = (_Float16)f.z; o.h[3] = (_Float16)f.w;
  reinterpret_cast<ushort4*>(out)[i] = o.u;
}

// ---------------------------------------------------------------------------
// W_enc [KD][DSAE] fp32  ->  Wt [DSAE][KD] f16   (32x32 LDS-tiled transpose)
// Makes the GEMM B-tile K-contiguous: no per-element LDS transpose in the
// inner loop, and makes the tile a plain 2D strided block for the TDM.
// ---------------------------------------------------------------------------
__global__ __launch_bounds__(256) void cvt_transpose_f32_f16(const float* __restrict__ in,
                                                             _Float16* __restrict__ out) {
  __shared__ _Float16 tile[32][33];
  const int n0 = blockIdx.x * 32;   // DSAE block
  const int k0 = blockIdx.y * 32;   // KD block
  const int tr = threadIdx.x >> 5;  // 0..7
  const int tc = threadIdx.x & 31;
#pragma unroll
  for (int i = 0; i < 4; ++i) {
    const int r = tr + i * 8;
    tile[r][tc] = (_Float16)in[(size_t)(k0 + r) * DSAE + n0 + tc];
  }
  __syncthreads();
#pragma unroll
  for (int i = 0; i < 4; ++i) {
    const int r = tr + i * 8;
    out[(size_t)(n0 + r) * KD + k0 + tc] = tile[tc][r];
  }
}

// ---------------------------------------------------------------------------
// c0[s] = b_enc[s] - sum_r b_dec[r] * W_enc[r, s]   (reads transposed Wt rows)
// ---------------------------------------------------------------------------
__global__ __launch_bounds__(256) void sae_enc_bias(const _Float16* __restrict__ Wt,
                                                    const float* __restrict__ b_enc,
                                                    const float* __restrict__ b_dec,
                                                    float* __restrict__ c0) {
  const int s = blockIdx.x * 256 + threadIdx.x;
  float c = b_enc[s];
  const _Float16* wrow = Wt + (size_t)s * KD;
  for (int r = 0; r < KD; r += 8) {
    union { uint4 u; _Float16 h[8]; } q;
    q.u = *reinterpret_cast<const uint4*>(wrow + r);
#pragma unroll
    for (int j = 0; j < 8; ++j) c -= b_dec[r + j] * (float)q.h[j];
  }
  c0[s] = c;
}

// ---------------------------------------------------------------------------
// Tensor Data Mover: 2D tile [128 rows x 32 f16] -> LDS with HW padding so the
// LDS row stride is 40 f16 (pad_interval=16 DWORDs, pad_amount=4 DWORDs).
// D# bitfields per CDNA5 ISA ch.8 (groups 0/1; groups 2/3 zero => <=2D use).
// ---------------------------------------------------------------------------
#if USE_TDM
__device__ __forceinline__ unsigned lds_off(const void* p) {
  return (unsigned)(unsigned long long)(uintptr_t)p;   // low 32 bits = LDS addr
}

__device__ __forceinline__ void tdm_load_tile_f16(unsigned lds_addr,
                                                  const _Float16* gptr,
                                                  unsigned tensor_d0,
                                                  unsigned tensor_d1,
                                                  unsigned long long stride_e) {
  const unsigned long long ga = (unsigned long long)(uintptr_t)gptr;
  u32x4 g0;
  g0[0] = 1u;                                              // count=1, user mode
  g0[1] = lds_addr;                                        // lds_addr [63:32]
  g0[2] = (unsigned)(ga & 0xffffffffull);                  // global_addr lo
  g0[3] = (unsigned)((ga >> 32) & 0x01ffffffull)           // global_addr hi (57b)
        | (2u << 30);                                      // type=2 ("image")
  i32x8 g1;
  g1[0] = (1 << 16)        // data_size = 2 bytes
        | (1 << 20)        // pad_enable
        | (3 << 22)        // pad_interval code 3 = 16 DWORDs (one 32-f16 row)
        | (3 << 25);       // pad_amount  code 3 = 4 DWORDs (8 f16) -> stride 40
  g1[1] = (int)((tensor_d0 & 0xffffu) << 16);                        // dim0 lo16
  g1[2] = (int)((tensor_d0 >> 16) | ((tensor_d1 & 0xffffu) << 16));  // dim0 hi / dim1 lo
  g1[3] = (int)((tensor_d1 >> 16) | (32u << 16));                    // dim1 hi / tile_dim0=32
  g1[4] = 128;                                                       // tile_dim1=128, tile_dim2=0
  g1[5] = (int)(stride_e & 0xffffffffull);                           // dim0_stride lo32
  g1[6] = (int)((stride_e >> 32) & 0xffffull);                       // dim0_stride hi16
  g1[7] = 0;
  const i32x4 z4 = {0, 0, 0, 0};
#if __clang_major__ >= 23
  const i32x8 z8 = {0, 0, 0, 0, 0, 0, 0, 0};
  __builtin_amdgcn_tensor_load_to_lds(g0, g1, z4, z4, z8, 0);
#else
  __builtin_amdgcn_tensor_load_to_lds(g0, g1, z4, z4, 0);
#endif
}
#endif  // USE_TDM

// ---------------------------------------------------------------------------
// Encode GEMM: pre[M=4096, N=16384] = x_h[M,K] * Wt[N,K]^T + c0[N]
// 128x128x32 block tile, 256 threads (8 waves), 2x4 16x16 tiles per wave.
// TDM path: double-buffered tensor_load_to_lds, one barrier per K-step.
// ---------------------------------------------------------------------------
#define BM  128
#define BN  128
#define BKT 32
#define LDA 40   // BKT + 8 f16 pad (matches TDM pad descriptor)

constexpr int NBUF = USE_TDM ? 2 : 1;

__global__ __launch_bounds__(256) void sae_encode_gemm(const _Float16* __restrict__ A,
                                                       const _Float16* __restrict__ Bt,
                                                       const float* __restrict__ c0,
                                                       float* __restrict__ pre) {
  __shared__ __align__(16) _Float16 As[NBUF][BM][LDA];
  __shared__ __align__(16) _Float16 Bs[NBUF][BN][LDA];

  const int tid  = threadIdx.x;
  const int n0   = blockIdx.x * BN;
  const int m0   = blockIdx.y * BM;
  const int wave = tid >> 5;
  const int lane = tid & 31;
  const int half = lane >> 4;         // K-half selector (ISA 16-bit layout)
  const int lm   = lane & 15;         // row (A) / col (B) within 16x16 tile
  const int wm0  = (wave >> 1) * 32;  // 4 waves along M
  const int wn0  = (wave & 1) * 64;   // 2 waves along N

  v8f acc[2][4] = {};

  // Fragment layout (ISA 7.12.2): lane half h holds K = h*8+0..7 and
  // 16+h*8+0..7 -> two contiguous 16-byte LDS reads per fragment.
  auto compute_step = [&](int bi) {
    union F16V { v16h v; uint4 u[2]; };
    F16V af[2], bf[4];
#pragma unroll
    for (int mt = 0; mt < 2; ++mt) {
      const int m = wm0 + mt * 16 + lm;
      af[mt].u[0] = *reinterpret_cast<const uint4*>(&As[bi][m][half * 8]);
      af[mt].u[1] = *reinterpret_cast<const uint4*>(&As[bi][m][16 + half * 8]);
    }
#pragma unroll
    for (int nt = 0; nt < 4; ++nt) {
      const int n = wn0 + nt * 16 + lm;
      bf[nt].u[0] = *reinterpret_cast<const uint4*>(&Bs[bi][n][half * 8]);
      bf[nt].u[1] = *reinterpret_cast<const uint4*>(&Bs[bi][n][16 + half * 8]);
    }
#pragma unroll
    for (int mt = 0; mt < 2; ++mt)
#pragma unroll
      for (int nt = 0; nt < 4; ++nt)
        acc[mt][nt] = __builtin_amdgcn_wmma_f32_16x16x32_f16(
            false, af[mt].v, false, bf[nt].v, (short)0, acc[mt][nt], false, false);
  };

#if USE_TDM
  const bool issuer = (wave == 0);
  const _Float16* Abase = A  + (size_t)m0 * KD;
  const _Float16* Bbase = Bt + (size_t)n0 * KD;
  if (issuer) {
    tdm_load_tile_f16(lds_off(&As[0][0][0]), Abase, KD, BATCH, KD);
    tdm_load_tile_f16(lds_off(&Bs[0][0][0]), Bbase, KD, DSAE, KD);
  }
  int buf = 0;
  for (int k0 = 0; k0 < KD; k0 += BKT) {
    if (issuer) __builtin_amdgcn_s_wait_tensorcnt(0);   // step-k0 tiles landed
    __syncthreads();                                    // publish LDS to all waves
    if (issuer && (k0 + BKT < KD)) {                    // prefetch next K-step
      tdm_load_tile_f16(lds_off(&As[buf ^ 1][0][0]), Abase + k0 + BKT, KD, BATCH, KD);
      tdm_load_tile_f16(lds_off(&Bs[buf ^ 1][0][0]), Bbase + k0 + BKT, KD, DSAE, KD);
    }
    compute_step(buf);                                  // overlaps with TDM DMA
    buf ^= 1;
  }
#else
  // Fallback: manual vectorized staging (transpose-free thanks to Wt layout)
  const int arow = tid >> 1;
  const int acol = (tid & 1) * 16;
  for (int k0 = 0; k0 < KD; k0 += BKT) {
    if (k0 + BKT < KD) {
      __builtin_prefetch(A  + (size_t)(m0 + arow) * KD + k0 + BKT + acol, 0, 1);
      __builtin_prefetch(Bt + (size_t)(n0 + arow) * KD + k0 + BKT + acol, 0, 1);
    }
    const uint4* sa = reinterpret_cast<const uint4*>(A  + (size_t)(m0 + arow) * KD + k0 + acol);
    const uint4* sb = reinterpret_cast<const uint4*>(Bt + (size_t)(n0 + arow) * KD + k0 + acol);
    uint4 a0 = sa[0], a1 = sa[1], b0 = sb[0], b1 = sb[1];
    *reinterpret_cast<uint4*>(&As[0][arow][acol])     = a0;
    *reinterpret_cast<uint4*>(&As[0][arow][acol + 8]) = a1;
    *reinterpret_cast<uint4*>(&Bs[0][arow][acol])     = b0;
    *reinterpret_cast<uint4*>(&Bs[0][arow][acol + 8]) = b1;
    __syncthreads();
    compute_step(0);
    __syncthreads();
  }
#endif

  // Epilogue: C/D layout — VGPR r: lanes0-15 -> M=r, lanes16-31 -> M=8+r, N=lm
#pragma unroll
  for (int nt = 0; nt < 4; ++nt) {
    const int gn = n0 + wn0 + nt * 16 + lm;
    const float cb = c0[gn];
#pragma unroll
    for (int mt = 0; mt < 2; ++mt)
#pragma unroll
      for (int r = 0; r < 8; ++r) {
        const int gm = m0 + wm0 + mt * 16 + half * 8 + r;
        pre[(size_t)gm * DSAE + gn] = acc[mt][nt][r] + cb;
      }
  }
}

// ---------------------------------------------------------------------------
// TopK(32) + ReLU, in-place on z (which holds pre on entry).
// ---------------------------------------------------------------------------
__global__ __launch_bounds__(256) void sae_topk(float* __restrict__ z,
                                                int* __restrict__ tki,
                                                float* __restrict__ tkv) {
  const int b = blockIdx.x;
  const int t = threadIdx.x;
  float* rowp = z + (size_t)b * DSAE;

  float v[64];
#pragma unroll
  for (int i = 0; i < 64; ++i) v[i] = rowp[t + (i << 8)];
#pragma unroll
  for (int i = 0; i < 64; ++i) rowp[t + (i << 8)] = 0.0f;   // own cols only

  float mv = -3.4e38f; int mi = 0;
#pragma unroll
  for (int i = 0; i < 64; ++i)
    if (v[i] > mv) { mv = v[i]; mi = t + (i << 8); }

  __shared__ float lmax[256];
  __shared__ int   larg[256];
  __shared__ int   si[TOPK];
  __shared__ float sv[TOPK];

  for (int r = 0; r < TOPK; ++r) {
    lmax[t] = mv; larg[t] = mi;
    __syncthreads();
    for (int s = 128; s > 0; s >>= 1) {
      if (t < s && lmax[t + s] > lmax[t]) { lmax[t] = lmax[t + s]; larg[t] = larg[t + s]; }
      __syncthreads();
    }
    const int wi = larg[0];
    const float wv = lmax[0];
    if (t == 0) { si[r] = wi; sv[r] = wv > 0.f ? wv : 0.f; }
    __syncthreads();
    if (t == (wi & 255)) {          // winner removes element, rescans registers
      const int slot = wi >> 8;
      mv = -3.4e38f; mi = t;
#pragma unroll
      for (int i = 0; i < 64; ++i) {
        if (i == slot) v[i] = -3.4e38f;
        if (v[i] > mv) { mv = v[i]; mi = t + (i << 8); }
      }
    }
    __syncthreads();
  }

  if (t < TOPK) {
    rowp[si[t]] = sv[t];
    tki[b * TOPK + t] = si[t];
    tkv[b * TOPK + t] = sv[t];
  }
}

// ---------------------------------------------------------------------------
// Sparse decode: x_hat[b,:] = sum_j val_j * W_dec[idx_j,:] + b_dec, plus loss.
// ---------------------------------------------------------------------------
__global__ __launch_bounds__(256) void sae_decode(const int* __restrict__ tki,
                                                  const float* __restrict__ tkv,
                                                  const _Float16* __restrict__ Wd,
                                                  const float* __restrict__ b_dec,
                                                  const float* __restrict__ x,
                                                  float* __restrict__ xhat,
                                                  float* __restrict__ loss) {
  const int b = blockIdx.x, t = threadIdx.x;
  __shared__ int   si[TOPK];
  __shared__ float sv[TOPK];
  __shared__ float red[256];
  if (t < TOPK) { si[t] = tki[b * TOPK + t]; sv[t] = tkv[b * TOPK + t]; }
  __syncthreads();

  float acc[16] = {};
  for (int j = 0; j < TOPK; ++j) {
    const _Float16* wrow = Wd + (size_t)si[j] * KD;
    const float vj = sv[j];
#pragma unroll
    for (int e = 0; e < 16; ++e)
      acc[e] += vj * (float)wrow[t + (e << 8)];
  }

  float lsum = 0.f;
#pragma unroll
  for (int e = 0; e < 16; ++e) {
    const int c = t + (e << 8);
    const float xv = acc[e] + b_dec[c];
    xhat[(size_t)b * KD + c] = xv;
    const float d = xv - x[(size_t)b * KD + c];
    lsum += d * d;
  }
  red[t] = lsum;
  __syncthreads();
  for (int s = 128; s > 0; s >>= 1) {
    if (t < s) red[t] += red[t + s];
    __syncthreads();
  }
  if (t == 0) atomicAdd(loss, red[0] * (1.0f / 32768.0f));  // mean over B*T
}

// ---------------------------------------------------------------------------
extern "C" void kernel_launch(void* const* d_in, const int* in_sizes, int n_in,
                              void* d_out, int out_size, void* d_ws, size_t ws_size,
                              hipStream_t stream) {
  const float* x     = (const float*)d_in[0];   // [B, KD]
  const float* W_enc = (const float*)d_in[1];   // [KD, DSAE]
  const float* b_enc = (const float*)d_in[2];   // [DSAE]
  const float* W_dec = (const float*)d_in[3];   // [DSAE, KD]
  const float* b_dec = (const float*)d_in[4];   // [KD]
  // d_in[5] = k (==32, hardcoded)

  float* out  = (float*)d_out;
  float* loss = out;                                  // 1
  float* xhat = out + 1;                              // B*KD
  float* z    = out + 1 + (size_t)BATCH * KD;         // B*DSAE (pre-act scratch)

  // Workspace (~169 MB). Wt buffer reused: W_enc^T(f16) then W_dec(f16).
  char* w = (char*)d_ws;
  _Float16* Wt  = (_Float16*)(w);                       // 134,217,728 B
  _Float16* xh  = (_Float16*)(w + 134217728);           //  33,554,432 B
  float*    c0  = (float*)   (w + 167772160);           //      65,536 B
  int*      tki = (int*)     (w + 167837696);           //     524,288 B
  float*    tkv = (float*)   (w + 168361984);           //     524,288 B

  const int NW = KD * DSAE;      // 67,108,864
  const int NX = BATCH * KD;     // 16,777,216

  hipMemsetAsync(loss, 0, sizeof(float), stream);

  // 1) conversions: W_enc -> f16 transposed [DSAE][KD] (L2-resident: 134 MB)
  dim3 tgrid(DSAE / 32, KD / 32);
  cvt_transpose_f32_f16<<<tgrid, 256, 0, stream>>>(W_enc, Wt);
  cvt_f32_f16<<<NX / 4 / 256, 256, 0, stream>>>(x, xh, NX / 4);

  // 2) fused encoder bias  c0 = b_enc - b_dec @ W_enc
  sae_enc_bias<<<DSAE / 256, 256, 0, stream>>>(Wt, b_enc, b_dec, c0);

  // 3) encode GEMM -> pre (written into z region of d_out)
  dim3 ggrid(DSAE / BN, BATCH / BM);
  sae_encode_gemm<<<ggrid, 256, 0, stream>>>(xh, Wt, c0, z);

  // 4) TopK+ReLU in place on z; compact (idx, val) for decode
  sae_topk<<<BATCH, 256, 0, stream>>>(z, tki, tkv);

  // 5) W_dec -> f16 (reuse Wt buffer; stream order guarantees GEMM is done)
  cvt_f32_f16<<<NW / 4 / 256, 256, 0, stream>>>(W_dec, Wt, NW / 4);

  // 6) sparse decode + reconstruction loss
  sae_decode<<<BATCH, 256, 0, stream>>>(tki, tkv, Wt, b_dec, x, xhat, loss);
}
